// EncoderCoreDecoder_14577119003009
// MI455X (gfx1250) — matrix operand order repaired and checked by reading, not compiled, not run on hardware
//
#include <hip/hip_runtime.h>
#include <hip/hip_bf16.h>
#include <initializer_list>
#include <stdint.h>

typedef __attribute__((ext_vector_type(16))) _Float16 v16h;
typedef __attribute__((ext_vector_type(8)))  _Float16 v8h;
typedef __attribute__((ext_vector_type(8)))  float    v8f;

#define N_V 50000
#define N_E 800000
#define N_U 64
#define HID 128
#define LAT 32
#define KMAX 256        // max concat input width (core edge MLP)
#define MLP_WAVES 4     // waves per block; each wave owns a 16-row tile

// ---------------- arg structs (shared host/device) ----------------
// is_half=1 segments are f16 activations with width==32, stride==32 (64B rows).
struct Seg { const void* ptr; const int* idx; int width; int stride; int is_half; };
struct MlpArgs {
  Seg seg[8]; int nseg; int ktiles; int rows;
  const _Float16* w0; const float* b0;   // w0: column-major f16 [128][Kpad]
  const _Float16* w1; const float* b1;   // w1: column-major f16 [32][128]
  const float* gamma; const float* beta;
  _Float16* out;                         // [rows][32] f16 activations
};

__device__ __forceinline__ uint32_t lds_byte_off(const void* p) {
  // CDNA5 flat aperture: the LDS offset is the low 32 bits of the flat address (ISA 10.2).
  return (uint32_t)(uint64_t)p;
}

// ---------------- fused gather + MLP(K->128->32) + ReLU + LayerNorm ----------
__global__ __launch_bounds__(MLP_WAVES * 32)
void mlp16_ln_kernel(MlpArgs A) {
  __shared__ __align__(32) _Float16 lds[MLP_WAVES * (16 * KMAX + 16 * HID)];
  const int wave = threadIdx.x >> 5, lane = threadIdx.x & 31;
  const int m = lane & 15, g = lane >> 4;   // A row / B,C column = m; lane group = g
  _Float16* Ain = lds + wave * (16 * KMAX + 16 * HID);
  _Float16* Hid = Ain + 16 * KMAX;
  const int Kpad = A.ktiles * 32;
  const long base = ((long)blockIdx.x * MLP_WAVES + wave) * 16;

  // ---- stage 16-row A tile into LDS ----
  int off = 0;
  for (int s = 0; s < A.nseg; ++s) {
    const Seg sg = A.seg[s];
    if (sg.is_half) {
      // f16 activation segment (64B rows): async global->LDS copy, gather folded
      // into per-lane addresses. 4 x B64 per lane covers the 16x64B tile.
      const char* gp = (const char*)sg.ptr;
      const int boff = (lane & 7) * 8;
      const uint32_t lbase = lds_byte_off(Ain) + off * 2 + boff;
      for (int i = 0; i < 4; ++i) {
        const int r = i * 4 + (lane >> 3);
        long row = base + r; if (row >= A.rows) row = 0;   // clamp (rows masked at store)
        const long srow = sg.idx ? (long)sg.idx[row] : row;
        const uint64_t gaddr = (uint64_t)(gp + srow * 64 + boff);
        const uint32_t laddr = lbase + (uint32_t)(r * Kpad * 2);
        asm volatile("global_load_async_to_lds_b64 %0, %1, off"
                     :: "v"(laddr), "v"(gaddr) : "memory");
      }
    } else if (sg.width == 32) {
      // f32 segment, width 32: two rows/iter, float2 load, packed b32 DS store.
      const float* fp = (const float*)sg.ptr;
      const int c2 = (lane & 15) * 2;
      for (int i = 0; i < 8; ++i) {
        const int r = i * 2 + g;
        const long row = base + r;
        float vx = 0.f, vy = 0.f;
        if (row < A.rows) {
          const long sr = sg.idx ? (long)sg.idx[row] : row;
          const float2 v = *(const float2*)(fp + sr * (long)sg.stride + c2);
          vx = v.x; vy = v.y;
        }
        union { _Float16 h[2]; uint32_t u; } pk;
        pk.h[0] = (_Float16)vx; pk.h[1] = (_Float16)vy;
        *(uint32_t*)&Ain[r * Kpad + off + c2] = pk.u;
      }
    } else {
      // f32 segment, width 16 (encoder globals only)
      const float* fp = (const float*)sg.ptr;
      for (int t = lane; t < 16 * 16; t += 32) {
        const int r = t >> 4, c = t & 15;
        const long row = base + r;
        float v = 0.f;
        if (row < A.rows) {
          const long sr = sg.idx ? (long)sg.idx[row] : row;
          v = fp[sr * (long)sg.stride + c];
        }
        Ain[r * Kpad + off + c] = (_Float16)v;
      }
    }
    off += sg.width;
  }
  if (off < Kpad) {            // zero-pad columns up to K tile boundary
    const int pw = Kpad - off;
    for (int t = lane; t < 16 * pw; t += 32) {
      const int r = t / pw, c = t - r * pw;
      Ain[r * Kpad + off + c] = (_Float16)0.f;
    }
  }
  asm volatile("s_wait_asynccnt 0x0" ::: "memory");
  __syncthreads();

  union HV { v16h v; v8h h[2]; };

  // ---- layer 1: [16 x Kpad] @ [Kpad x 128] ----
  v8f hacc[8];
  for (int nt = 0; nt < 8; ++nt) {
    v8f acc = {};
    for (int kt = 0; kt < A.ktiles; ++kt) {
      HV af, bf;
      const _Float16* ap = Ain + m * Kpad + kt * 32 + g * 8;   // K = kt*32+g*8+[0..7], +16
      af.h[0] = *(const v8h*)ap;
      af.h[1] = *(const v8h*)(ap + 16);
      const _Float16* bp = A.w0 + (size_t)(nt * 16 + m) * Kpad + kt * 32 + g * 8;
      bf.h[0] = *(const v8h*)bp;
      bf.h[1] = *(const v8h*)(bp + 16);
      acc = __builtin_amdgcn_wmma_f32_16x16x32_f16(false, af.v, false, bf.v,
                                                   (short)0, acc, false, false);
    }
    const float bias = A.b0[nt * 16 + m];
    for (int r = 0; r < 8; ++r) { float v = acc[r] + bias; acc[r] = v > 0.f ? v : 0.f; }
    hacc[nt] = acc;
  }
  // C-layout -> A-layout handoff through LDS
  for (int nt = 0; nt < 8; ++nt)
    for (int r = 0; r < 8; ++r)
      Hid[(r + g * 8) * HID + nt * 16 + m] = (_Float16)hacc[nt][r];
  __syncthreads();

  // ---- layer 2: [16 x 128] @ [128 x 32] ----
  v8f o0 = {}, o1 = {};
  for (int kt = 0; kt < 4; ++kt) {
    HV af, b0f, b1f;
    const _Float16* ap = Hid + m * HID + kt * 32 + g * 8;
    af.h[0] = *(const v8h*)ap;
    af.h[1] = *(const v8h*)(ap + 16);
    const _Float16* q0 = A.w1 + (size_t)(m)      * HID + kt * 32 + g * 8;
    const _Float16* q1 = A.w1 + (size_t)(16 + m) * HID + kt * 32 + g * 8;
    b0f.h[0] = *(const v8h*)q0; b0f.h[1] = *(const v8h*)(q0 + 16);
    b1f.h[0] = *(const v8h*)q1; b1f.h[1] = *(const v8h*)(q1 + 16);
    o0 = __builtin_amdgcn_wmma_f32_16x16x32_f16(false, af.v, false, b0f.v, (short)0, o0, false, false);
    o1 = __builtin_amdgcn_wmma_f32_16x16x32_f16(false, af.v, false, b1f.v, (short)0, o1, false, false);
  }

  // ---- bias + ReLU + LayerNorm(32) + f16 store ----
  const float c0 = A.b1[m],     c1 = A.b1[16 + m];
  const float g0 = A.gamma[m],  g1 = A.gamma[16 + m];
  const float be0 = A.beta[m],  be1 = A.beta[16 + m];
  for (int r = 0; r < 8; ++r) {
    float a0 = o0[r] + c0; a0 = a0 > 0.f ? a0 : 0.f;
    float a1 = o1[r] + c1; a1 = a1 > 0.f ? a1 : 0.f;
    float s = a0 + a1, q = a0 * a0 + a1 * a1;
    s += __shfl_xor(s, 1, 16); q += __shfl_xor(q, 1, 16);
    s += __shfl_xor(s, 2, 16); q += __shfl_xor(q, 2, 16);
    s += __shfl_xor(s, 4, 16); q += __shfl_xor(q, 4, 16);
    s += __shfl_xor(s, 8, 16); q += __shfl_xor(q, 8, 16);
    const float mean = s * (1.f / 32.f);
    float var = q * (1.f / 32.f) - mean * mean;
    var = var < 0.f ? 0.f : var;
    const float rstd = rsqrtf(var + 1e-5f);
    const long row = base + r + g * 8;
    if (row < A.rows) {
      A.out[row * 32 + m]      = (_Float16)((a0 - mean) * rstd * g0 + be0);
      A.out[row * 32 + 16 + m] = (_Float16)((a1 - mean) * rstd * g1 + be1);
    }
  }
}

// ---------------- helper kernels ----------------
__global__ void zero_u32_kernel(uint32_t* p, long n) {
  long t = (long)blockIdx.x * blockDim.x + threadIdx.x;
  if (t < n) p[t] = 0u;
}

// f32 [K][O] row-major -> f16 [O][Kpad] column-major (zero-padded K)
__global__ void conv_w_kernel(const float* W, _Float16* out, int K, int Kpad, int O) {
  int t = blockIdx.x * blockDim.x + threadIdx.x;
  if (t >= O * Kpad) return;
  int o = t / Kpad, k = t - o * Kpad;
  out[(size_t)o * Kpad + k] = (k < K) ? (_Float16)W[(size_t)k * O + o] : (_Float16)0.f;
}

// agg[idx[r]][c] += src[r][c]   (f16 src, unsorted receiver index, f32 accum)
__global__ void scatter_add32_kernel(const _Float16* src, const int* idx, float* dst, long rows) {
  long t = (long)blockIdx.x * blockDim.x + threadIdx.x;
  if (t >= rows * 32) return;
  long r = t >> 5; int c = (int)(t & 31);
  atomicAdd(&dst[(long)idx[r] * 32 + c], (float)src[t]);
}

// run-length segment sum for SORTED idx (f16 src); also counts rows per segment
__global__ void segsum32_kernel(const _Float16* x, const int* idx, float* sums, float* cnt, long rows) {
  const int c = threadIdx.x & 31, sub = threadIdx.x >> 5;  // 8 sub-groups of 32
  const long r0 = ((long)blockIdx.x * 8 + sub) * 128;
  float acc = 0.f, cacc = 0.f; int cur = -1;
  for (int i = 0; i < 128; ++i) {
    const long r = r0 + i;
    if (r >= rows) break;
    const int s = idx[r];
    if (s != cur) {
      if (cur >= 0) { atomicAdd(&sums[(long)cur * 32 + c], acc); if (c == 0) atomicAdd(&cnt[cur], cacc); }
      cur = s; acc = 0.f; cacc = 0.f;
    }
    acc += (float)x[r * 32 + c]; cacc += 1.f;
  }
  if (cur >= 0) { atomicAdd(&sums[(long)cur * 32 + c], acc); if (c == 0) atomicAdd(&cnt[cur], cacc); }
}

__global__ void segdiv_kernel(float* sums, const float* cnt, int nseg) {
  int t = blockIdx.x * blockDim.x + threadIdx.x;
  if (t >= nseg * 32) return;
  sums[t] /= fmaxf(cnt[t >> 5], 1.f);
}

// out[r][o] = b[o] + sum_k x[r][k] * W[k][o]   (f16 activations, f32 result)
__global__ void linear_out_kernel(const _Float16* x, const float* W, const float* b,
                                  float* out, long rows, int O) {
  long t = (long)blockIdx.x * blockDim.x + threadIdx.x;
  if (t >= rows * O) return;
  long r = t / O; int o = (int)(t - r * O);
  float acc = b[o];
  const _Float16* xr = x + r * 32;
  for (int k = 0; k < 32; ++k) acc += (float)xr[k] * W[k * O + o];
  out[t] = acc;
}

// ---------------- host orchestration ----------------
static inline int cdiv(long a, long b) { return (int)((a + b - 1) / b); }

extern "C" void kernel_launch(void* const* d_in, const int* in_sizes, int n_in,
                              void* d_out, int out_size, void* d_ws, size_t ws_size,
                              hipStream_t stream) {
  (void)in_sizes; (void)n_in; (void)out_size; (void)ws_size;
  const float* x   = (const float*)d_in[0];
  const int*   ei  = (const int*)d_in[1];
  const float* ea  = (const float*)d_in[2];
  const float* u   = (const float*)d_in[3];
  const int*   vix = (const int*)d_in[4];
  const int*   eix = (const int*)d_in[5];
  const int* row = ei; const int* col = ei + N_E;

  // params flattened depth-first in dict insertion order:
  // enc{node,edge,glob}, core{edge,node,glob}, dec{node,edge,glob}, v_out, e_out, u_out
  const int Ks[9] = {64, 32, 16, 256, 160, 128, 32, 32, 32};

  char* ws = (char*)d_ws; size_t off = 0;
  auto alloc = [&](size_t bytes) -> void* {
    off = (off + 255) & ~(size_t)255;
    void* p = ws + off; off += bytes; return p;
  };
  // f16 activation state (entire graph state ~115MB -> L2 resident)
  _Float16* x0 = (_Float16*)alloc((size_t)N_V * 32 * 2);
  _Float16* xl = (_Float16*)alloc((size_t)N_V * 32 * 2);
  _Float16* e0 = (_Float16*)alloc((size_t)N_E * 32 * 2);
  _Float16* el = (_Float16*)alloc((size_t)N_E * 32 * 2);
  _Float16* u0 = (_Float16*)alloc((size_t)N_U * 32 * 2);
  _Float16* ul = (_Float16*)alloc((size_t)N_U * 32 * 2);
  // f32 aggregation buffers (atomic accumulation)
  float* aggE  = (float*)alloc((size_t)N_V * 32 * 4);
  float* aggV  = (float*)alloc((size_t)N_U * 32 * 4);
  float* aggE2 = (float*)alloc((size_t)N_U * 32 * 4);
  float* cntV  = (float*)alloc((size_t)N_U * 4);
  float* cntE  = (float*)alloc((size_t)N_U * 4);

  struct MlpW { const float *b0, *b1, *g, *b; _Float16 *w0, *w1; int ktiles; } M[9];
  for (int i = 0; i < 9; ++i) {
    const int bi = 6 + 6 * i;
    const float* W0 = (const float*)d_in[bi + 0];
    M[i].b0 = (const float*)d_in[bi + 1];
    const float* W1 = (const float*)d_in[bi + 2];
    M[i].b1 = (const float*)d_in[bi + 3];
    M[i].g  = (const float*)d_in[bi + 4];
    M[i].b  = (const float*)d_in[bi + 5];
    const int K = Ks[i], Kpad = ((K + 31) / 32) * 32;
    M[i].ktiles = Kpad / 32;
    M[i].w0 = (_Float16*)alloc((size_t)Kpad * HID * 2);
    M[i].w1 = (_Float16*)alloc((size_t)HID * LAT * 2);
    conv_w_kernel<<<cdiv((long)HID * Kpad, 256), 256, 0, stream>>>(W0, M[i].w0, K, Kpad, HID);
    conv_w_kernel<<<cdiv((long)LAT * HID, 256), 256, 0, stream>>>(W1, M[i].w1, HID, HID, LAT);
  }

  auto zero = [&](void* p, long n32) { zero_u32_kernel<<<cdiv(n32, 256), 256, 0, stream>>>((uint32_t*)p, n32); };
  auto mlp = [&](int i, std::initializer_list<Seg> segs, _Float16* out, long rows) {
    MlpArgs A{}; int k = 0;
    for (const Seg& s : segs) A.seg[k++] = s;
    A.nseg = k; A.ktiles = M[i].ktiles; A.rows = (int)rows;
    A.w0 = M[i].w0; A.b0 = M[i].b0; A.w1 = M[i].w1; A.b1 = M[i].b1;
    A.gamma = M[i].g; A.beta = M[i].b; A.out = out;
    mlp16_ln_kernel<<<cdiv(rows, 16 * MLP_WAVES), 32 * MLP_WAVES, 0, stream>>>(A);
  };

  // zero initial recurrent state (f16 => n halfs / 2 dwords)
  zero(xl, (long)N_V * 16); zero(el, (long)N_E * 16); zero(ul, (long)N_U * 16);

  // ---- encoder (f32 inputs -> f16 latents) ----
  mlp(0, {{x,      nullptr, 32, 64, 0}, {x + 32, nullptr, 32, 64, 0}}, x0, N_V);
  mlp(1, {{ea,     nullptr, 32, 32, 0}}, e0, N_E);
  mlp(2, {{u,      nullptr, 16, 16, 0}}, u0, N_U);

  // ---- core (2 steps, in-place state update) ----
  for (int step = 0; step < 2; ++step) {
    mlp(3, {{x0, row, 32, 32, 1}, {xl, row, 32, 32, 1},
            {x0, col, 32, 32, 1}, {xl, col, 32, 32, 1},
            {e0, nullptr, 32, 32, 1}, {el, nullptr, 32, 32, 1},
            {u0, eix, 32, 32, 1}, {ul, eix, 32, 32, 1}}, el, N_E);
    zero(aggE, (long)N_V * 32);
    scatter_add32_kernel<<<cdiv((long)N_E * 32, 256), 256, 0, stream>>>(el, col, aggE, N_E);
    mlp(4, {{x0, nullptr, 32, 32, 1}, {xl, nullptr, 32, 32, 1},
            {aggE, nullptr, 32, 32, 0},
            {u0, vix, 32, 32, 1}, {ul, vix, 32, 32, 1}}, xl, N_V);
    zero(aggV, (long)N_U * 32); zero(cntV, N_U);
    segsum32_kernel<<<cdiv(N_V, 8 * 128), 256, 0, stream>>>(xl, vix, aggV, cntV, N_V);
    segdiv_kernel<<<cdiv((long)N_U * 32, 256), 256, 0, stream>>>(aggV, cntV, N_U);
    zero(aggE2, (long)N_U * 32); zero(cntE, N_U);
    segsum32_kernel<<<cdiv(N_E, 8 * 128), 256, 0, stream>>>(el, eix, aggE2, cntE, N_E);
    segdiv_kernel<<<cdiv((long)N_U * 32, 256), 256, 0, stream>>>(aggE2, cntE, N_U);
    mlp(5, {{u0, nullptr, 32, 32, 1}, {ul, nullptr, 32, 32, 1},
            {aggV, nullptr, 32, 32, 0}, {aggE2, nullptr, 32, 32, 0}}, ul, N_U);
  }

  // ---- decoder (in place) ----
  mlp(6, {{xl, nullptr, 32, 32, 1}}, xl, N_V);
  mlp(7, {{el, nullptr, 32, 32, 1}}, el, N_E);
  mlp(8, {{ul, nullptr, 32, 32, 1}}, ul, N_U);

  // ---- output linears ----
  float* outN = (float*)d_out;
  float* outE = outN + (size_t)N_V * 2;
  float* outU = outE + (size_t)N_E * 1;
  linear_out_kernel<<<cdiv((long)N_V * 2, 256), 256, 0, stream>>>(
      xl, (const float*)d_in[60], (const float*)d_in[61], outN, N_V, 2);
  linear_out_kernel<<<cdiv((long)N_E, 256), 256, 0, stream>>>(
      el, (const float*)d_in[62], (const float*)d_in[63], outE, N_E, 1);
  linear_out_kernel<<<cdiv((long)N_U, 256), 256, 0, stream>>>(
      ul, (const float*)d_in[64], (const float*)d_in[65], outU, N_U, 1);
}
